// GCN2_BP_23055384445767
// MI455X (gfx1250) — matrix operand order repaired and checked by reading, not compiled, not run on hardware
//
#include <hip/hip_runtime.h>
#include <cmath>

typedef __attribute__((ext_vector_type(16))) __bf16          v16bf;
typedef __attribute__((ext_vector_type(16))) unsigned short  v16us;
typedef __attribute__((ext_vector_type(8)))  float           v8f;
typedef __attribute__((ext_vector_type(8)))  unsigned int    v8u;

constexpr int HD   = 64;    // hidden width
constexpr int FD   = 256;   // input features
constexpr int NCLS = 40;    // classes

// ---------------- helpers ----------------

__device__ __forceinline__ unsigned short f2bf(float f) {
  unsigned u = __builtin_bit_cast(unsigned, f);
  u += 0x7FFFu + ((u >> 16) & 1u);          // round-to-nearest-even
  return (unsigned short)(u >> 16);
}

__device__ __forceinline__ unsigned pk_bf16(float lo, float hi) {
  return (unsigned)f2bf(lo) | ((unsigned)f2bf(hi) << 16);
}

__device__ __forceinline__ v8f wmma_bf16(v16bf a, v16bf b, v8f c) {
  // v_wmma_f32_16x16x32_bf16  (neg_a, A, neg_b, B, c_mod, C, reuse_a, reuse_b)
  return __builtin_amdgcn_wmma_f32_16x16x32_bf16(false, a, false, b, (short)0, c,
                                                 false, false);
}

// A fragment (16x32 bf16) straight from a bf16 row (no conversion):
// lane holds row m = lane&15; halves K = lo+0..7 then 16+lo+0..7, lo = (lane/16)*8.
__device__ __forceinline__ v16bf a_frag_bf16(const unsigned short* __restrict__ row) {
  const int lo = ((threadIdx.x >> 4) & 1) * 8;
  uint4 a = *(const uint4*)(row + lo);
  uint4 b = *(const uint4*)(row + 16 + lo);
  v8u u = {a.x, a.y, a.z, a.w, b.x, b.y, b.z, b.w};
  return __builtin_bit_cast(v16bf, u);
}

// A fragment from an f32 row (converted in registers; round-1 codegen form).
__device__ __forceinline__ v16bf a_frag_f32(const float* __restrict__ p) {
  const int lo = ((threadIdx.x >> 4) & 1) * 8;
  v16us r;
#pragma unroll
  for (int i = 0; i < 8; ++i) {
    r[i]     = f2bf(p[lo + i]);
    r[i + 8] = f2bf(p[16 + lo + i]);
  }
  return __builtin_bit_cast(v16bf, r);
}

// B fragment (32x16 bf16): wt_row = K-contiguous (transposed) weight row for
// column n = lane&15, already offset by kstep*32.  halves K = kb+0..7 then
// 16+kb+0..7, kb = (lane/16)*8.
__device__ __forceinline__ v16bf b_frag(const unsigned short* wt_row) {
  const int kb = ((threadIdx.x >> 4) & 1) * 8;
  uint4 lo = *(const uint4*)(wt_row + kb);
  uint4 hi = *(const uint4*)(wt_row + 16 + kb);
  v8u u = {lo.x, lo.y, lo.z, lo.w, hi.x, hi.y, hi.z, hi.w};
  return __builtin_bit_cast(v16bf, u);
}

// ---------------- kernels ----------------

// x [N,256] f32 -> bf16 copy (coalesced, one pass).
__global__ void __launch_bounds__(256) k_prep_x(const float* __restrict__ x,
                                                unsigned* __restrict__ xb,
                                                long long ndw) {
  long long i = (long long)blockIdx.x * 256 + threadIdx.x;
  if (i >= ndw) return;
  float2 v = ((const float2*)x)[i];
  xb[i] = pk_bf16(v.x, v.y);
}

// W2 [4096,40] f32 -> W2rT [2560,64] bf16 ; W2rT[jc][k] = W2[(k*64+j)*40+c]
__global__ void __launch_bounds__(256) k_prep_w2(const float* __restrict__ W2,
                                                 unsigned* __restrict__ W2T32) {
  int i = blockIdx.x * 256 + threadIdx.x;          // 2560*32 dwords
  if (i >= (HD * NCLS) * (HD / 2)) return;
  int jc = i >> 5, d = i & 31;                     // 32 dwords per 64-half row
  int j = jc / NCLS, c = jc - NCLS * j;
  W2T32[i] = pk_bf16(W2[((2 * d) * HD + j) * NCLS + c],
                     W2[((2 * d + 1) * HD + j) * NCLS + c]);
}

// h0 = relu(x @ W0 + b0) ; also seeds h = h0.  A comes from bf16 x-copy.
__global__ void __launch_bounds__(256) k_gemm0(const unsigned short* __restrict__ xb,
                                               const float* __restrict__ W0,
                                               const float* __restrict__ b0,
                                               float* __restrict__ h0,
                                               float* __restrict__ h,
                                               int ntiles) {
  __shared__ __align__(16) unsigned short sW[HD * FD];   // W0^T bf16 [n][k], 32KB
  unsigned* sW32 = (unsigned*)sW;
  for (int i = threadIdx.x; i < HD * (FD / 2); i += 256) {
    int n = i >> 7, d = i & 127;                   // 128 dwords per 256-half row
    sW32[i] = pk_bf16(W0[(2 * d) * HD + n], W0[(2 * d + 1) * HD + n]);
  }
  __syncthreads();
  const int tile = blockIdx.x * 8 + (threadIdx.x >> 5);
  if (tile >= ntiles) return;
  const int lane = threadIdx.x & 31;
  const int mn   = lane & 15;
  const unsigned short* xrow = xb + ((long long)tile * 16 + mn) * FD;

  v8f zero = {};
  v8f acc[4] = {zero, zero, zero, zero};
#pragma unroll
  for (int ks = 0; ks < FD / 32; ++ks) {
    v16bf a = a_frag_bf16(xrow + ks * 32);         // two global b128 loads
#pragma unroll
    for (int ct = 0; ct < 4; ++ct)
      acc[ct] = wmma_bf16(a, b_frag(sW + (ct * 16 + mn) * FD + ks * 32), acc[ct]);
  }
  const int hi = lane >> 4;
#pragma unroll
  for (int ct = 0; ct < 4; ++ct) {
    float bias = b0[ct * 16 + mn];
#pragma unroll
    for (int v = 0; v < 8; ++v) {
      int mm = v + 8 * hi;                       // C layout: M = v + 8*(lane/16)
      long long r = (long long)tile * 16 + mm;
      float val = fmaxf(acc[ct][v] + bias, 0.f);
      h0[r * HD + ct * 16 + mn] = val;
      h [r * HD + ct * 16 + mn] = val;
    }
  }
}

// agg[dst] += w * h[src] : one thread per (edge, float4) -> b128 gathers,
// 4 L2-resident f32 atomics each (segment-sum semantics preserved).
__global__ void __launch_bounds__(256) k_spmm(const int* __restrict__ ei,
                                              const float* __restrict__ ew,
                                              const float* __restrict__ h,
                                              float* __restrict__ agg,
                                              long long E) {
  long long gid = (long long)blockIdx.x * 256 + threadIdx.x;
  if (gid >= E * (HD / 4)) return;
  long long e = gid >> 4;
  int f4 = (int)(gid & 15);
  int s = ei[e], d = ei[E + e];
  float w = ew[e];
  float4 v = ((const float4*)(h + (long long)s * HD))[f4];
  float* a = agg + (long long)d * HD + f4 * 4;
  unsafeAtomicAdd(a + 0, w * v.x);
  unsafeAtomicAdd(a + 1, w * v.y);
  unsafeAtomicAdd(a + 2, w * v.z);
  unsafeAtomicAdd(a + 3, w * v.w);
}

// h = relu((1-beta)*hmix + beta*(hmix @ Wl)),  hmix = (1-alpha)*agg + alpha*h0
__global__ void __launch_bounds__(256) k_layer(const float* __restrict__ agg,
                                               const float* __restrict__ h0,
                                               float* __restrict__ h,
                                               const float* __restrict__ Wl,  // [64][64] f32
                                               float alpha, float beta, int ntiles) {
  __shared__ __align__(16) unsigned short sW[HD * HD];  // Wl^T bf16 [n][k], 8KB
  __shared__ __align__(16) float sH[8][16 * HD];        // per-wave hmix tile, 32KB
  unsigned* sW32 = (unsigned*)sW;
  for (int i = threadIdx.x; i < HD * (HD / 2); i += 256) {
    int n = i >> 5, d = i & 31;                   // 32 dwords per 64-half row
    sW32[i] = pk_bf16(Wl[(2 * d) * HD + n], Wl[(2 * d + 1) * HD + n]);
  }
  const int wave = threadIdx.x >> 5, lane = threadIdx.x & 31;
  const int tile = blockIdx.x * 8 + wave;
  float* tl = sH[wave];
  if (tile < ntiles) {
    const int r = lane & 15, cb = (lane >> 4) * 32;
    long long rowg = (long long)tile * 16 + r;
    const float4* pa = (const float4*)(agg + rowg * HD + cb);
    const float4* p0 = (const float4*)(h0 + rowg * HD + cb);
    float4* pt = (float4*)(tl + r * HD + cb);
#pragma unroll
    for (int i = 0; i < 8; ++i) {
      float4 a4 = pa[i], g4 = p0[i], m4;
      m4.x = (1.f - alpha) * a4.x + alpha * g4.x;
      m4.y = (1.f - alpha) * a4.y + alpha * g4.y;
      m4.z = (1.f - alpha) * a4.z + alpha * g4.z;
      m4.w = (1.f - alpha) * a4.w + alpha * g4.w;
      pt[i] = m4;
    }
  }
  __syncthreads();
  if (tile >= ntiles) return;
  const int mn = lane & 15, hi = lane >> 4;
  v16bf a0 = a_frag_f32(tl + mn * HD);
  v16bf a1 = a_frag_f32(tl + mn * HD + 32);
  v8f zero = {};
  v8f acc[4] = {zero, zero, zero, zero};
#pragma unroll
  for (int ct = 0; ct < 4; ++ct) {
    acc[ct] = wmma_bf16(a0, b_frag(sW + (ct * 16 + mn) * HD),      acc[ct]);
    acc[ct] = wmma_bf16(a1, b_frag(sW + (ct * 16 + mn) * HD + 32), acc[ct]);
  }
#pragma unroll
  for (int ct = 0; ct < 4; ++ct)
#pragma unroll
    for (int v = 0; v < 8; ++v) {
      int mm = v + 8 * hi;
      float hm  = tl[mm * HD + ct * 16 + mn];
      float val = fmaxf((1.f - beta) * hm + beta * acc[ct][v], 0.f);
      h[((long long)tile * 16 + mm) * HD + ct * 16 + mn] = val;
    }
}

// Fused head: u = h @ W2r (WMMA), logits[m][c] += u[m][j*40+c]*h[m][j] in LDS,
// then log_softmax.  Never materializes the [N,4096] outer product.
__global__ void __launch_bounds__(256) k_head(const float* __restrict__ h,
                                              const unsigned short* __restrict__ W2T,
                                              const float* __restrict__ b2,
                                              float* __restrict__ out) {
  __shared__ __align__(16) float sH[16 * HD];   // h tile, 4KB
  __shared__ float sL[16][NCLS];                // logits accumulator
  const int tile = blockIdx.x;
  ((float4*)sH)[threadIdx.x] =
      ((const float4*)(h + (long long)tile * 16 * HD))[threadIdx.x];
  for (int i = threadIdx.x; i < 16 * NCLS; i += 256)
    ((float*)sL)[i] = b2[i % NCLS];
  __syncthreads();

  const int wave = threadIdx.x >> 5, lane = threadIdx.x & 31;
  const int mn = lane & 15, hi = lane >> 4;
  v16bf a0 = a_frag_f32(sH + mn * HD);
  v16bf a1 = a_frag_f32(sH + mn * HD + 32);
  for (int jt = wave; jt < (HD * NCLS) / 16; jt += 8) {   // 160 column tiles
    const unsigned short* wrow = W2T + (jt * 16 + mn) * HD;
    // prefetch next iteration's B row into cache (global_prefetch_b8)
    __builtin_prefetch(W2T + ((jt + 8) * 16 + mn) * HD, 0, 0);
    v8f acc = {};
    acc = wmma_bf16(a0, b_frag(wrow),      acc);
    acc = wmma_bf16(a1, b_frag(wrow + 32), acc);
#pragma unroll
    for (int v = 0; v < 8; ++v) {
      int mm = v + 8 * hi;
      int jc = jt * 16 + mn;
      int j = jc / NCLS, c = jc - NCLS * j;
      unsafeAtomicAdd(&sL[mm][c], acc[v] * sH[mm * HD + j]);  // ds atomic add
    }
  }
  __syncthreads();
  if (threadIdx.x < 16) {
    int m = threadIdx.x;
    float mx = -1e30f;
    for (int c = 0; c < NCLS; ++c) mx = fmaxf(mx, sL[m][c]);
    float s = 0.f;
    for (int c = 0; c < NCLS; ++c) s += __expf(sL[m][c] - mx);
    float lg = mx + __logf(s);
    long long rowg = (long long)tile * 16 + m;
    for (int c = 0; c < NCLS; ++c) out[rowg * NCLS + c] = sL[m][c] - lg;
  }
}

// ---------------- host launcher ----------------

extern "C" void kernel_launch(void* const* d_in, const int* in_sizes, int n_in,
                              void* d_out, int out_size, void* d_ws, size_t ws_size,
                              hipStream_t stream) {
  const float* x  = (const float*)d_in[0];
  const int*   ei = (const int*)  d_in[1];
  const float* ew = (const float*)d_in[2];
  const float* W0 = (const float*)d_in[3];
  const float* b0 = (const float*)d_in[4];
  const float* Wl = (const float*)d_in[5];
  const float* W2 = (const float*)d_in[6];
  const float* b2 = (const float*)d_in[7];

  const int       N      = in_sizes[0] / FD;    // 50000
  const long long E      = in_sizes[2];         // 1.6M
  const int       ntiles = N / 16;              // 3125 (exact)

  // workspace: h0 | h | agg (f32 N*64 each) | W2rT bf16 [2560][64] | x_bf16 [N][256]
  float* h0  = (float*)d_ws;
  float* h   = h0  + (size_t)N * HD;
  float* agg = h   + (size_t)N * HD;
  unsigned short* W2T = (unsigned short*)(agg + (size_t)N * HD);
  unsigned short* xb  = W2T + (size_t)HD * NCLS * HD;

  long long xdw = (long long)N * FD / 2;
  k_prep_x<<<(unsigned)((xdw + 255) / 256), 256, 0, stream>>>(x, (unsigned*)xb, xdw);
  k_prep_w2<<<((HD * NCLS * HD / 2) + 255) / 256, 256, 0, stream>>>(W2, (unsigned*)W2T);
  k_gemm0<<<(ntiles + 7) / 8, 256, 0, stream>>>(xb, W0, b0, h0, h, ntiles);

  for (int l = 0; l < 8; ++l) {
    hipMemsetAsync(agg, 0, (size_t)N * HD * sizeof(float), stream);
    long long work = E * (HD / 4);
    k_spmm<<<(unsigned)((work + 255) / 256), 256, 0, stream>>>(ei, ew, h, agg, E);
    float beta = logf(0.5f / (float)(l + 1) + 1.0f);
    k_layer<<<(ntiles + 7) / 8, 256, 0, stream>>>(agg, h0, h, Wl + l * HD * HD,
                                                  0.1f, beta, ntiles);
  }
  k_head<<<ntiles, 256, 0, stream>>>(h, W2T, b2, (float*)d_out);
}
